// AtomicMNISTClassifier_30064771072596
// MI455X (gfx1250) — compile-verified, matrix operand order checked by reading, not run on hardware
//
#include <hip/hip_runtime.h>
#include <hip/hip_bf16.h>

// --- WMMA vector types (CDNA5 / gfx1250, wave32) ---
typedef __attribute__((ext_vector_type(16))) __bf16 v16bf;
typedef __attribute__((ext_vector_type(8)))  float  v8f;

#define MT 128        // M tile per workgroup
#define NT 128        // N tile per workgroup
#define KT 32         // K step (bf16 WMMA K)
#define PITCH 40      // LDS row pitch in bf16 elems (32 data + 8 pad)

__device__ __forceinline__ v16bf lds_frag(const __bf16* p) {
    v16bf r;
    __builtin_memcpy(&r, p, 32);   // 2x ds_load_b128
    return r;
}

// ---- Unguarded full-16 staging (hot path, uniform control flow) ----
template<typename T>
__device__ __forceinline__ void stage16_full(const T* __restrict__ src, __bf16* dst);

template<>
__device__ __forceinline__ void stage16_full<float>(const float* __restrict__ src, __bf16* dst) {
    const float4* s4 = (const float4*)src;       // 4x global_load_b128
#pragma unroll
    for (int j = 0; j < 4; ++j) {
        float4 v = s4[j];
        dst[j*4+0] = (__bf16)v.x; dst[j*4+1] = (__bf16)v.y;
        dst[j*4+2] = (__bf16)v.z; dst[j*4+3] = (__bf16)v.w;
    }
}

template<>
__device__ __forceinline__ void stage16_full<__bf16>(const __bf16* __restrict__ src, __bf16* dst) {
    __builtin_memcpy(dst, src, 32);              // 2x b128 load + 2x ds_store_b128
}

// ---- Guarded tail staging (runs at most once per workgroup) ----
template<typename T>
__device__ __forceinline__ void stage16_tail(const T* __restrict__ src, int valid, __bf16* dst) {
#pragma unroll
    for (int i = 0; i < 16; ++i) {
        float f = (i < valid) ? (float)src[i] : 0.0f;
        dst[i] = (__bf16)f;
    }
}

// C[M,N] = act(A[M,K] @ W[N,K]^T + bias); bf16 MACs, fp32 accum, bf16 out.
// Block: 256 threads = 8 waves; each wave computes a 16(M) x 128(N) strip.
template<typename TIN, bool RELU>
__global__ __launch_bounds__(256) void wmma_gemm_bias_act(
    const TIN*    __restrict__ A,     // [M, K] row-major
    const __bf16* __restrict__ W,     // [N, K] row-major (bf16 weights)
    const float*  __restrict__ bias,  // [N]
    __bf16*       __restrict__ C,     // [M, N] row-major bf16
    int M, int N, int K)
{
    __shared__ __bf16 sA[2][MT][PITCH];
    __shared__ __bf16 sB[2][NT][PITCH];

    const int m0   = blockIdx.x * MT;
    const int n0   = blockIdx.y * NT;
    const int tid  = threadIdx.x;
    const int lane = tid & 31;
    const int wave = tid >> 5;

    // staging role: thread -> (row, K-half)
    const int srow  = tid >> 1;          // 0..127
    const int shalf = (tid & 1) * 16;    // 0 or 16

    const int nsteps = (K + KT - 1) / KT;

    auto stage_full = [&](int buf, int k0) {
        stage16_full<TIN>  (A + (size_t)(m0 + srow) * K + k0 + shalf, &sA[buf][srow][shalf]);
        stage16_full<__bf16>(W + (size_t)(n0 + srow) * K + k0 + shalf, &sB[buf][srow][shalf]);
    };
    auto stage_tail = [&](int buf, int k0) {
        int ka    = k0 + shalf;
        int valid = K - ka;
        if (valid < 0) valid = 0;
        stage16_tail<TIN>  (A + (size_t)(m0 + srow) * K + ka, valid, &sA[buf][srow][shalf]);
        stage16_tail<__bf16>(W + (size_t)(n0 + srow) * K + ka, valid, &sB[buf][srow][shalf]);
    };

    v8f acc[NT / 16];
#pragma unroll
    for (int i = 0; i < NT / 16; ++i)
#pragma unroll
        for (int j = 0; j < 8; ++j) acc[i][j] = 0.0f;

    // Prologue (K >= KT in all our layers; step 0 is always full)
    stage_full(0, 0);
    __syncthreads();

    // Fragment addressing (16-bit A & B layouts are symmetric):
    // lane L holds row (L&15), K-range [(L>>4)*16, +16) -> 32 contiguous bytes in LDS.
    const int fr = lane & 15;
    const int fk = (lane >> 4) * 16;

    int buf = 0;
    for (int s = 0; s < nsteps; ++s) {
        if (s + 1 < nsteps) {
            const int k1 = (s + 1) * KT;
            if (k1 + KT <= K) stage_full(buf ^ 1, k1);   // uniform branch: no divergence
            else              stage_tail(buf ^ 1, k1);   // taken once, at the K remainder
        }

        // Preload all fragments, then issue 8 independent WMMAs back-to-back
        // (single s_wait_dscnt instead of one per matrix op).
        v16bf a = lds_frag(&sA[buf][wave * 16 + fr][fk]);
        v16bf b[NT / 16];
#pragma unroll
        for (int n = 0; n < NT / 16; ++n)
            b[n] = lds_frag(&sB[buf][n * 16 + fr][fk]);
#pragma unroll
        for (int n = 0; n < NT / 16; ++n)
            acc[n] = __builtin_amdgcn_wmma_f32_16x16x32_bf16(
                false, a, false, b[n], (short)0, acc[n], false, false);

        __syncthreads();
        buf ^= 1;
    }

    // Epilogue: C/D layout: VGPR i, lanes 0-15 -> M=i, lanes 16-31 -> M=8+i; N = lane&15.
    const int col0 = lane & 15;
    const int rsel = (lane >> 4) * 8;
#pragma unroll
    for (int n = 0; n < NT / 16; ++n) {
        const int col = n0 + n * 16 + col0;
        const float bv = bias[col];
#pragma unroll
        for (int i = 0; i < 8; ++i) {
            const int row = m0 + wave * 16 + rsel + i;
            float v = acc[n][i] + bv;
            if (RELU) v = v > 0.0f ? v : 0.0f;
            C[(size_t)row * N + col] = (__bf16)v;
        }
    }
}

// Layer 3 (256 -> 10) + log_softmax, one wave per batch row.
__global__ __launch_bounds__(256) void head_logsoftmax(
    const __bf16* __restrict__ H,   // [M, 256] bf16
    const float*  __restrict__ W3,  // [10, 256] fp32 (L2-resident)
    const float*  __restrict__ b3,  // [10]
    float*        __restrict__ out, // [M, 10]
    int M)
{
    const int gtid = blockIdx.x * blockDim.x + threadIdx.x;
    const int row  = gtid >> 5;
    const int lane = threadIdx.x & 31;
    if (row >= M) return;

    float hv[8];
    {
        __bf16 tmp[8];
        __builtin_memcpy(tmp, H + (size_t)row * 256 + lane * 8, 16); // b128, coalesced
#pragma unroll
        for (int i = 0; i < 8; ++i) hv[i] = (float)tmp[i];
    }

    float acc[10];
#pragma unroll
    for (int n = 0; n < 10; ++n) {
        const float4* wr = (const float4*)(W3 + n * 256 + lane * 8);
        float4 w0 = wr[0], w1v = wr[1];
        float s = hv[0]*w0.x + hv[1]*w0.y + hv[2]*w0.z + hv[3]*w0.w
                + hv[4]*w1v.x + hv[5]*w1v.y + hv[6]*w1v.z + hv[7]*w1v.w;
#pragma unroll
        for (int off = 16; off >= 1; off >>= 1)
            s += __shfl_xor(s, off, 32);
        acc[n] = s + b3[n];
    }

    if (lane == 0) {
        float mx = acc[0];
#pragma unroll
        for (int n = 1; n < 10; ++n) mx = fmaxf(mx, acc[n]);
        float sum = 0.0f;
#pragma unroll
        for (int n = 0; n < 10; ++n) sum += __expf(acc[n] - mx);
        const float lse = mx + __logf(sum);
        float* o = out + (size_t)row * 10;
#pragma unroll
        for (int n = 0; n < 10; ++n) o[n] = acc[n] - lse;
    }
}

__global__ void cvt_f32_to_bf16(const float* __restrict__ s, __bf16* __restrict__ d, int n) {
    int i = blockIdx.x * blockDim.x + threadIdx.x;
    if (i < n) d[i] = (__bf16)s[i];
}

extern "C" void kernel_launch(void* const* d_in, const int* in_sizes, int n_in,
                              void* d_out, int out_size, void* d_ws, size_t ws_size,
                              hipStream_t stream) {
    (void)n_in; (void)out_size; (void)ws_size;

    const float* x  = (const float*)d_in[0];
    const float* w1 = (const float*)d_in[1];
    const float* b1 = (const float*)d_in[2];
    const float* w2 = (const float*)d_in[3];
    const float* b2 = (const float*)d_in[4];
    const float* w3 = (const float*)d_in[5];
    const float* b3 = (const float*)d_in[6];
    float* out = (float*)d_out;

    const int K1 = 784, N1 = 128, N2 = 256;
    const int M = in_sizes[0] / K1;   // 32768

    // Workspace layout (~25.5 MB): bf16 weights + bf16 activations.
    char* ws = (char*)d_ws;
    size_t off = 0;
    __bf16* w1b = (__bf16*)(ws + off); off += (size_t)N1 * K1 * 2; off = (off + 255) & ~(size_t)255;
    __bf16* w2b = (__bf16*)(ws + off); off += (size_t)N2 * N1 * 2; off = (off + 255) & ~(size_t)255;
    __bf16* h1  = (__bf16*)(ws + off); off += (size_t)M  * N1 * 2; off = (off + 255) & ~(size_t)255;
    __bf16* h2  = (__bf16*)(ws + off);

    cvt_f32_to_bf16<<<(N1 * K1 + 255) / 256, 256, 0, stream>>>(w1, w1b, N1 * K1);
    cvt_f32_to_bf16<<<(N2 * N1 + 255) / 256, 256, 0, stream>>>(w2, w2b, N2 * N1);

    // Layer 1: [M,784] fp32 (converted on the fly) @ [128,784]^T -> relu -> bf16
    wmma_gemm_bias_act<float,  true><<<dim3(M / MT, N1 / NT), 256, 0, stream>>>(x,  w1b, b1, h1, M, N1, K1);
    // Layer 2: [M,128] bf16 @ [256,128]^T -> relu -> bf16
    wmma_gemm_bias_act<__bf16, true><<<dim3(M / MT, N2 / NT), 256, 0, stream>>>(h1, w2b, b2, h2, M, N2, N1);
    // Layer 3 + log_softmax: one wave per row
    head_logsoftmax<<<(M * 32 + 255) / 256, 256, 0, stream>>>(h2, w3, b3, out, M);
}
